// DirectionalFilterBank_47553877901539
// MI455X (gfx1250) — matrix operand order repaired and checked by reading, not compiled, not run on hardware
//
#include <hip/hip_runtime.h>

// CDNA5 / gfx1250 fused shear + depthwise 9x9 conv via Toeplitz WMMA.
// D = A x B + C with V_WMMA_F32_16X16X32_F16 (wave32):
//   A[m][k]  = sheared input row, columns 16*m + k - 4   (k = 0..23, pad to 32)
//   B[k][n]  = w[dy][k - n]  (banded Toeplitz, zero outside 0..8)
//   D[m][n]  = output row pixel x = 16*m + n              (256 px per wave-row)

typedef __attribute__((ext_vector_type(16))) _Float16 v16h;
typedef __attribute__((ext_vector_type(8)))  _Float16 v8h;
typedef __attribute__((ext_vector_type(8)))  float    v8f;

#define HH   256
#define WW   256
#define CCH  64
#define BB   4
#define KF   9
#define ROWS 8            // output rows per block
#define SROWS (ROWS + 8)  // staged sheared rows (halo 4 each side)
#define SW   272          // 4 left zeros + 256 + 12 right zeros (16B-aligned stride)
#define BLOCK 128         // 4 waves

__global__ __launch_bounds__(BLOCK)
void dfb_shear_conv(const float* __restrict__ src, const float* __restrict__ filt,
                    float* __restrict__ dst, int sign)
{
    __shared__ _Float16 srow[SROWS][SW];   // sheared+padded input rows (f16)
    __shared__ _Float16 wz[KF][48];        // wz[dy][16+j] = w[dy][j], zeros elsewhere

    const int tid   = threadIdx.x;
    const int id    = blockIdx.x;          // b*2048 + c*32 + ytile
    const int ytile = id & 31;
    const int c     = (id >> 5) & (CCH - 1);
    const int b     = id >> 11;
    const int y0    = ytile * ROWS;

    // ---- stage zero-padded filter row bank ----
    for (int idx = tid; idx < KF * 48; idx += BLOCK) {
        const int dy = idx / 48, j = idx % 48;
        float v = 0.0f;
        if (j >= 16 && j <= 24) v = filt[(c * KF + dy) * KF + (j - 16)];
        wz[dy][j] = (_Float16)v;
    }

    // ---- stage sheared input rows (exact: frac(ix) == 0.5 for H=W=256) ----
    const float* base = src + (size_t)(b * CCH + c) * (HH * WW);
    for (int idx = tid; idx < SROWS * SW; idx += BLOCK) {
        const int rr   = idx / SW;
        const int col  = idx % SW;
        const int xcol = col - 4;           // sheared-image x
        const int r    = y0 - 4 + rr;       // image y
        float v = 0.0f;
        if (r >= 0 && r < HH && xcol >= 0 && xcol < WW) {
            const int ix0 = xcol + ((sign > 0) ? (r - 128) : (127 - r));
            const float v0 = (ix0 >= 0 && ix0 < WW)       ? base[r * WW + ix0]     : 0.0f;
            const float v1 = (ix0 + 1 >= 0 && ix0 + 1 < WW) ? base[r * WW + ix0 + 1] : 0.0f;
            v = 0.5f * (v0 + v1);
        }
        srow[rr][col] = (_Float16)v;
    }
    __syncthreads();

    // ---- per-wave: 2 output rows, one 16x16 D-tile == full 256-px row each ----
    const int wave  = tid >> 5;
    const int lane  = tid & 31;
    const int m     = lane & 15;                 // A row (x-block) / D column n
    const int abase = (lane & 16) ? 16 : 0;      // A k order used here (contiguous): K 0-15 vs 16-31
    const int n     = lane & 15;
    const int kbase = (lane & 16) ? 16 : 0;      // B k order used here (contiguous): K 0-15 vs 16-31

    v8f acc0 = {};
    v8f acc1 = {};
    const int r0 = 2 * wave;                     // LDS row of (y - 4) for first out row

    #pragma unroll
    for (int dy = 0; dy < KF; ++dy) {
        // Banded Toeplitz B: contiguous 16 halves of the padded filter row
        const _Float16* wp = &wz[dy][16 + kbase - n];
        v16h bm;
        #pragma unroll
        for (int i = 0; i < 16; ++i) bm[i] = wp[i];

        // A operands: two aligned 16B LDS loads per lane (contiguous K, matches B)
        union { v16h v; v8h h[2]; } a0, a1;
        const _Float16* p0 = &srow[r0 + dy][16 * m + abase];
        const _Float16* p1 = &srow[r0 + dy + 1][16 * m + abase];
        a0.h[0] = *(const v8h*)(p0);
        a0.h[1] = *(const v8h*)(p0 + 8);
        a1.h[0] = *(const v8h*)(p1);
        a1.h[1] = *(const v8h*)(p1 + 8);

        acc0 = __builtin_amdgcn_wmma_f32_16x16x32_f16(false, a0.v, false, bm,
                                                      (short)0, acc0, false, false);
        acc1 = __builtin_amdgcn_wmma_f32_16x16x32_f16(false, a1.v, false, bm,
                                                      (short)0, acc1, false, false);
    }

    // ---- store: D layout -> x = (v + (lane>=16 ? 8 : 0))*16 + n ----
    float* drow = dst + ((size_t)(b * CCH + c) * HH + (y0 + 2 * wave)) * WW;
    const int xoff = ((lane & 16) ? 128 : 0) + n;
    #pragma unroll
    for (int v = 0; v < 8; ++v) {
        drow[v * 16 + xoff]      = acc0[v];
        drow[WW + v * 16 + xoff] = acc1[v];
    }
}

extern "C" void kernel_launch(void* const* d_in, const int* in_sizes, int n_in,
                              void* d_out, int out_size, void* d_ws, size_t ws_size,
                              hipStream_t stream) {
    const float* x    = (const float*)d_in[0];   // (4,64,256,256) f32
    const float* filt = (const float*)d_in[1];   // (64,1,9,9) f32
    float* out = (float*)d_out;                  // (8,4,64,256,256) f32

    const size_t N = (size_t)BB * CCH * HH * WW; // one tensor
    float* tA = (float*)d_ws;                    // 4 scratch tensors
    float* tB = tA + N;
    float* tC = tB + N;
    float* tD = tC + N;

    dim3 grid(BB * CCH * (HH / ROWS));
    dim3 block(BLOCK);
    auto conv = [&](const float* s, float* d, int sign) {
        dfb_shear_conv<<<grid, block, 0, stream>>>(s, filt, d, sign);
    };

    // level 3
    conv(x, tA, +1);            // t1
    conv(x, tB, -1);            // t2
    // level 2
    conv(tA, tC, +1);           // t11
    conv(tA, tD, -1);           // t12   (tA free)
    conv(tB, tA, +1);           // t21   (reuse tA)
    conv(tB, out + 0 * N, -1);  // t22 -> temp in out[0]
    // leaves from t22 first (then out[0] can be reclaimed)
    conv(out + 0 * N, out + 6 * N, +1);  // leaf 6
    conv(out + 0 * N, out + 7 * N, -1);  // leaf 7
    // remaining leaves
    conv(tC, out + 0 * N, +1);  // leaf 0 (overwrites temp)
    conv(tC, out + 1 * N, -1);  // leaf 1
    conv(tD, out + 2 * N, +1);  // leaf 2
    conv(tD, out + 3 * N, -1);  // leaf 3
    conv(tA, out + 4 * N, +1);  // leaf 4
    conv(tA, out + 5 * N, -1);  // leaf 5
}
